// GraphSAGE_79061757985056
// MI455X (gfx1250) — matrix operand order repaired and checked by reading, not compiled, not run on hardware
//
#include <hip/hip_runtime.h>
#include <hip/hip_bf16.h>

typedef __attribute__((ext_vector_type(16))) __bf16 v16bf;
typedef __attribute__((ext_vector_type(8)))  __bf16 v8bf;
typedef __attribute__((ext_vector_type(8)))  float  v8f;

#define K_DIM 128

// ---------------------------------------------------------------------------
// fp32 -> bf16 weight conversion (run once per launch; weights are constant
// across all 6250 M-tiles, so convert 80K elements once instead of per-block).
// ---------------------------------------------------------------------------
__global__ void cvt_f32_bf16(const float* __restrict__ in,
                             __bf16* __restrict__ out, int n)
{
    const int i = blockIdx.x * blockDim.x + threadIdx.x;
    if (i < n) out[i] = (__bf16)in[i];
}

// ---------------------------------------------------------------------------
// Edge scatter: agg[dst] += feat[src] (and deg[dst] += 1 on the first pass).
// One wave per edge: 32 lanes x float4 = the full 512B feature row, perfectly
// coalesced. Working set (~170MB) is L2-resident on the 192MB L2; atomics
// resolve there. Prefetch the next edge's source row to hide the
// index-load -> gather dependency.
// ---------------------------------------------------------------------------
__global__ __launch_bounds__(256)
void sage_scatter(const float* __restrict__ feat,
                  const int* __restrict__ src,
                  const int* __restrict__ dst,
                  float* __restrict__ agg,
                  float* __restrict__ deg,
                  int nEdges, int computeDeg)
{
    const int gtid   = blockIdx.x * blockDim.x + threadIdx.x;
    const int wave   = gtid >> 5;
    const int lane   = threadIdx.x & 31;
    const int nWaves = (gridDim.x * blockDim.x) >> 5;

    for (int e = wave; e < nEdges; e += nWaves) {
        const int s = src[e];
        const int d = dst[e];

        const int eNext = e + nWaves;
        if (eNext < nEdges) {
            const int sN = src[eNext];
            __builtin_prefetch(feat + (size_t)sN * K_DIM + lane * 4, 0, 3);
        }

        const float4* fp = (const float4*)(feat + (size_t)s * K_DIM);
        const float4 v = fp[lane];
        float* ap = agg + (size_t)d * K_DIM + lane * 4;
        atomicAdd(ap + 0, v.x);
        atomicAdd(ap + 1, v.y);
        atomicAdd(ap + 2, v.z);
        atomicAdd(ap + 3, v.w);
        if (computeDeg && lane == 0) atomicAdd(deg + d, 1.0f);
    }
}

// ---------------------------------------------------------------------------
// A-fragment gather from an LDS bf16 row (16B ds_load_b128 x2).
// Contiguous k order used here (B matches): lane half h takes K={a1..a1+15}, a1 = kk + 16h.
// ---------------------------------------------------------------------------
__device__ __forceinline__ v16bf load_a_frag(const __bf16* row, int a1)
{
    const v8bf lo = *(const v8bf*)(row + a1);
    const v8bf hi = *(const v8bf*)(row + a1 + 8);
    return __builtin_shufflevector(lo, hi, 0, 1, 2, 3, 4, 5, 6, 7,
                                            8, 9, 10, 11, 12, 13, 14, 15);
}

// ---------------------------------------------------------------------------
// Fused SAGE GEMM: out = [relu]( mean(agg,deg) @ Wl^T + xroot @ Wr^T + bias )
// Block = 256 thr = 8 waves; one wave -> one 16x16 tile of the output block.
// Stage the shared 16-row M-tile once per block into LDS as bf16 (agg rows
// pre-scaled by rcp(max(deg,1))), then feed v_wmma_f32_16x16x32_bf16 from
// ds_load_b128 (A) and pre-converted bf16 global weights (B).
// ---------------------------------------------------------------------------
template<bool HAS_MEAN, bool RELU>
__global__ __launch_bounds__(256)
void sage_gemm(const float* __restrict__ xroot,
               const float* __restrict__ agg,
               const float* __restrict__ deg,
               const __bf16* __restrict__ Wlb,
               const __bf16* __restrict__ Wrb,
               const float* __restrict__ bias,
               float* __restrict__ out,
               int nNodes, int outDim)
{
    __shared__ __align__(16) __bf16 xs[16 * K_DIM];
    __shared__ __align__(16) __bf16 ms[16 * K_DIM];

    const int tid   = threadIdx.x;
    const int mBase = blockIdx.x << 4;

    // ---- cooperative stage: 16 rows x 128 cols, float4 per thread x2 ----
#pragma unroll
    for (int p = 0; p < 2; ++p) {
        const int flat = (p * 256 + tid) * 4;      // 0..2044
        const int row  = flat >> 7;
        const int col  = flat & 127;
        int gm = mBase + row;
        if (gm >= nNodes) gm = nNodes - 1;

        const float4 v = *(const float4*)(xroot + (size_t)gm * K_DIM + col);
        xs[flat + 0] = (__bf16)v.x;
        xs[flat + 1] = (__bf16)v.y;
        xs[flat + 2] = (__bf16)v.z;
        xs[flat + 3] = (__bf16)v.w;

        if (HAS_MEAN) {
            const float iv = __builtin_amdgcn_rcpf(fmaxf(deg[gm], 1.0f));
            const float4 a = *(const float4*)(agg + (size_t)gm * K_DIM + col);
            ms[flat + 0] = (__bf16)(a.x * iv);
            ms[flat + 1] = (__bf16)(a.y * iv);
            ms[flat + 2] = (__bf16)(a.z * iv);
            ms[flat + 3] = (__bf16)(a.w * iv);
        }
    }
    __syncthreads();

    const int nTiles = outDim >> 4;
    const int waveId = tid >> 5;
    if (waveId >= nTiles) return;               // head: 7 of 8 waves compute

    const int lane  = tid & 31;
    const int half  = lane >> 4;
    const int r     = lane & 15;
    const int nBase = waveId << 4;
    const int n     = nBase + r;

    v8f c;
    const float bv = bias[n];
#pragma unroll
    for (int i = 0; i < 8; ++i) c[i] = bv;

    const __bf16* xrow  = xs + r * K_DIM;
    const __bf16* mrowp = ms + r * K_DIM;
    const __bf16* wrrow = Wrb + (size_t)n * K_DIM;
    const __bf16* wlrow = HAS_MEAN ? (Wlb + (size_t)n * K_DIM) : nullptr;

#pragma unroll
    for (int kk = 0; kk < K_DIM; kk += 32) {
        const int a1 = kk + half * 16;           // A: lanes0-15 K 0-15 contig
        const int bb = kk + half * 16;           // B: lanes0-15 K 0-15 contig

        const v16bf aF = load_a_frag(xrow, a1);
        const v16bf bF = *(const v16bf*)(wrrow + bb);
        c = __builtin_amdgcn_wmma_f32_16x16x32_bf16(
                false, aF, false, bF, (short)0, c, false, false);

        if (HAS_MEAN) {
            const v16bf aM = load_a_frag(mrowp, a1);
            const v16bf bL = *(const v16bf*)(wlrow + bb);
            c = __builtin_amdgcn_wmma_f32_16x16x32_bf16(
                    false, aM, false, bL, (short)0, c, false, false);
        }
    }

    // C/D 16x16 f32 layout: vgpr i -> row mBase + half*8 + i, col n
    const int mrow = mBase + half * 8;
#pragma unroll
    for (int i = 0; i < 8; ++i) {
        const int row = mrow + i;
        if (row < nNodes) {
            float v = c[i];
            if (RELU) v = fmaxf(v, 0.0f);
            out[(size_t)row * outDim + n] = v;
        }
    }
}

// ---------------------------------------------------------------------------
extern "C" void kernel_launch(void* const* d_in, const int* in_sizes, int n_in,
                              void* d_out, int out_size, void* d_ws, size_t ws_size,
                              hipStream_t stream)
{
    const float* x    = (const float*)d_in[0];
    const int*   eidx = (const int*)  d_in[1];
    const float* Wl1  = (const float*)d_in[2];
    const float* bl1  = (const float*)d_in[3];
    const float* Wr1  = (const float*)d_in[4];
    const float* Wl2  = (const float*)d_in[5];
    const float* bl2  = (const float*)d_in[6];
    const float* Wr2  = (const float*)d_in[7];
    const float* Wout = (const float*)d_in[8];
    const float* bout = (const float*)d_in[9];

    const int N    = in_sizes[0] / K_DIM;   // 100000
    const int E    = in_sizes[1] / 2;       // 1600000
    const int OUTD = in_sizes[9];           // 112
    const int* src = eidx;                  // edge_index[0]
    const int* dst = eidx + E;              // edge_index[1]

    // workspace layout
    float* agg = (float*)d_ws;                    // N*128 f32
    float* h1  = agg + (size_t)N * K_DIM;         // N*128 f32
    float* h2  = h1  + (size_t)N * K_DIM;         // N*128 f32
    float* deg = h2  + (size_t)N * K_DIM;         // N f32
    __bf16* wlb1 = (__bf16*)(deg + N);            // 128*128 bf16 each
    __bf16* wrb1 = wlb1 + K_DIM * K_DIM;
    __bf16* wlb2 = wrb1 + K_DIM * K_DIM;
    __bf16* wrb2 = wlb2 + K_DIM * K_DIM;
    __bf16* wob  = wrb2 + K_DIM * K_DIM;          // OUTD*128 bf16
    float* outp  = (float*)d_out;

    const int mTiles = (N + 15) / 16;
    const dim3 sBlk(256), sGrd(2048);
    const int wE = K_DIM * K_DIM;

    // ---- one-time (per launch) weight conversion to bf16 ----
    cvt_f32_bf16<<<(wE + 255) / 256, 256, 0, stream>>>(Wl1, wlb1, wE);
    cvt_f32_bf16<<<(wE + 255) / 256, 256, 0, stream>>>(Wr1, wrb1, wE);
    cvt_f32_bf16<<<(wE + 255) / 256, 256, 0, stream>>>(Wl2, wlb2, wE);
    cvt_f32_bf16<<<(wE + 255) / 256, 256, 0, stream>>>(Wr2, wrb2, wE);
    cvt_f32_bf16<<<(OUTD * K_DIM + 255) / 256, 256, 0, stream>>>(Wout, wob, OUTD * K_DIM);

    // ---- layer 1 ----
    hipMemsetAsync(agg, 0, (size_t)N * K_DIM * sizeof(float), stream);
    hipMemsetAsync(deg, 0, (size_t)N * sizeof(float), stream);
    sage_scatter<<<sGrd, sBlk, 0, stream>>>(x, src, dst, agg, deg, E, 1);
    sage_gemm<true, true><<<mTiles, 256, 0, stream>>>(
        x, agg, deg, wlb1, wrb1, bl1, h1, N, K_DIM);

    // ---- layer 2 (deg reused) ----
    hipMemsetAsync(agg, 0, (size_t)N * K_DIM * sizeof(float), stream);
    sage_scatter<<<sGrd, sBlk, 0, stream>>>(h1, src, dst, agg, deg, E, 0);
    sage_gemm<true, true><<<mTiles, 256, 0, stream>>>(
        h1, agg, deg, wlb2, wrb2, bl2, h2, N, K_DIM);

    // ---- output head ----
    sage_gemm<false, false><<<mTiles, 256, 0, stream>>>(
        h2, nullptr, nullptr, nullptr, wob, bout, outp, N, OUTD);
}
